// NFReductionModel_85263690760584
// MI455X (gfx1250) — matrix operand order repaired and checked by reading, not compiled, not run on hardware
//
#include <hip/hip_runtime.h>

// ---------------------------------------------------------------------------
// NF reduction model on gfx1250 (MI455X).
// Compute-bound (~365 GFLOP of small GEMMs, all data L2-resident) =>
// v_wmma_f32_16x16x32_f16 with f32 accumulation.
// Rep flow: 4 waves / workgroup cooperate on one 16-row M-tile (N-split),
// activations + phi staged in LDS; ~4-5 waves/SIMD resident to hide WMMA
// and transcendental latency. nt-loop unrolled x2 for dual WMMA chains.
// ---------------------------------------------------------------------------

typedef __attribute__((ext_vector_type(16))) _Float16 v16h;
typedef __attribute__((ext_vector_type(8)))  float    v8f;

#define DEV __device__ __forceinline__

// ---------------- problem constants ----------------
constexpr int   BINS    = 8;
constexpr float BOUND   = 5.0f;
constexpr int   PDIM    = 3 * BINS - 1;    // 23
constexpr int   BATCH   = 16384;
constexpr int   X_DIM   = 32;
constexpr int   LATENT  = 8;
constexpr float L2PI    = 1.8378770664093453f;

// ---------------- workspace layout (dword units) ----------------
constexpr int FRAG_DW   = 256;             // one 16x16x32 B-fragment: 32 lanes * 8 dwords
constexpr int REP_FRAGS = 8 + 32 + 184;    // W1(1x8) + W2(4x8) + W3(4x46) = 224
constexpr int Z_FRAG0   = 3 * REP_FRAGS;   // 672
constexpr int Z_FRAGS   = 16 + 32 + 48;    // W1(2x8) + W2(4x8) + W3(4x12, N padded 184->192)
constexpr int ENC_W1    = Z_FRAG0 + 3 * Z_FRAGS;  // 960
constexpr int ENC_W2    = ENC_W1 + 8;             // 968
constexpr int ENC_W3    = ENC_W2 + 32;            // 1000
constexpr int TOT_FRAGS = ENC_W3 + 8;             // 1008

constexpr int BIAS0  = TOT_FRAGS * FRAG_DW;       // 258048 (float index into ws)
constexpr int REP_B0 = BIAS0;                     // per t: b1[128], b2[128], b3[736] => 992
constexpr int ZB0    = BIAS0 + 3 * 992;           // per t: b1[128], b2[128], b3[184] => 440
constexpr int EB0    = ZB0 + 3 * 440;             // b1[128], b2[128], b3[32]
constexpr int U_OFF  = 262656;                    // u buffer: BATCH*32 floats
constexpr int LD_OFF = U_OFF + BATCH * X_DIM;     // logdet: BATCH floats

// ---------------- 16B helper (POD so it can live in unions) ----------------
struct alignas(16) U4 { unsigned x, y, z, w; };

DEV v16h ld_bfrag(const unsigned* __restrict__ f, int lane) {
    union { v16h h; U4 u[2]; } t;
    const U4* q = reinterpret_cast<const U4*>(f + lane * 8);
    t.u[0] = q[0];
    t.u[1] = q[1];
    return t.h;
}

// A fragment (16x32 f16) from an LDS row-major buffer.
// lane<16 holds K 0..7 & 16..23 of row (lane&15); lane>=16 holds K 8..15 & 24..31.
DEV v16h ld_afrag(const _Float16* __restrict__ hb, int stride, int kt, int lane) {
    const int r  = lane & 15;
    const int hi = lane >> 4;
    const _Float16* base = hb + r * stride + kt * 32;
    union { v16h h; U4 u[2]; } t;
    t.u[0] = *reinterpret_cast<const U4*>(base + 8 * hi);
    t.u[1] = *reinterpret_cast<const U4*>(base + 16 + 8 * hi);
    return t.h;
}

// ---------------- WMMA MLP layers ----------------
// Work is split across the waves of the workgroup along the N-tile dimension:
// wave `wid` handles nt = wid, wid+NW, ...  ACT: 0 = none, 1 = relu, 2 = silu.
// Unconditional prefetch: one-past-the-end addresses still land in the
// workspace (bias region follows the fragment region), prefetch has no
// side effects, and the unguarded form avoids exec-mask churn per nt step.
template <int KT, int NT, int NW, int ACT>
DEV void mlp_layer_h(const _Float16* __restrict__ inb, int strideIn,
                     _Float16* __restrict__ outb, int strideOut, int colOff,
                     const unsigned* __restrict__ w, const float* __restrict__ bias,
                     int Nreal, int tid) {
    const int lane = tid & 31;
    const int wid  = tid >> 5;
    const int r  = lane & 15;
    const int hi = lane >> 4;
#pragma unroll 2
    for (int nt = wid; nt < NT; nt += NW) {
        __builtin_prefetch((const void*)(w + (size_t)(nt + NW) * KT * FRAG_DW), 0, 3);
        v8f c = {};
#pragma unroll
        for (int kt = 0; kt < KT; ++kt) {
            v16h a = ld_afrag(inb, strideIn, kt, lane);
            v16h b = ld_bfrag(w + (size_t)(nt * KT + kt) * FRAG_DW, lane);
            c = __builtin_amdgcn_wmma_f32_16x16x32_f16(false, a, false, b,
                                                       (short)0, c, false, false);
        }
        const int n  = nt * 16 + r;
        const float bb = (n < Nreal) ? bias[n] : 0.f;
#pragma unroll
        for (int v = 0; v < 8; ++v) {
            float val = c[v] + bb;
            if (ACT == 1) val = fmaxf(val, 0.f);
            if (ACT == 2) val = val / (1.f + __expf(-val));
            outb[(v + 8 * hi) * strideOut + colOff + n] = (_Float16)val;
        }
    }
}

// f32 output (final phi layer): bias add only.
template <int KT, int NT, int NW>
DEV void mlp_layer_f(const _Float16* __restrict__ inb, int strideIn,
                     float* __restrict__ outb, int strideOut,
                     const unsigned* __restrict__ w, const float* __restrict__ bias,
                     int Nreal, int tid) {
    const int lane = tid & 31;
    const int wid  = tid >> 5;
    const int r  = lane & 15;
    const int hi = lane >> 4;
#pragma unroll 2
    for (int nt = wid; nt < NT; nt += NW) {
        __builtin_prefetch((const void*)(w + (size_t)(nt + NW) * KT * FRAG_DW), 0, 3);
        v8f c = {};
#pragma unroll
        for (int kt = 0; kt < KT; ++kt) {
            v16h a = ld_afrag(inb, strideIn, kt, lane);
            v16h b = ld_bfrag(w + (size_t)(nt * KT + kt) * FRAG_DW, lane);
            c = __builtin_amdgcn_wmma_f32_16x16x32_f16(false, a, false, b,
                                                       (short)0, c, false, false);
        }
        const int n  = nt * 16 + r;
        const float bb = (n < Nreal) ? bias[n] : 0.f;
#pragma unroll
        for (int v = 0; v < 8; ++v)
            outb[(v + 8 * hi) * strideOut + n] = c[v] + bb;
    }
}

// ---------------- RQS spline (branch-free, register-resident) ----------------
DEV void rqs_knots(const float* __restrict__ phi, float* xk, float* yk, float* dk) {
    float mw = phi[0], mh = phi[8];
#pragma unroll
    for (int i = 1; i < 8; ++i) { mw = fmaxf(mw, phi[i]); mh = fmaxf(mh, phi[8 + i]); }
    float ew[8], eh[8], sw = 0.f, sh = 0.f;
#pragma unroll
    for (int i = 0; i < 8; ++i) {
        ew[i] = __expf(phi[i] - mw);      sw += ew[i];
        eh[i] = __expf(phi[8 + i] - mh);  sh += eh[i];
    }
    const float iw = 1.f / sw, ih = 1.f / sh;
    float cw = 0.f, ch = 0.f;
    xk[0] = -BOUND; yk[0] = -BOUND;
#pragma unroll
    for (int i = 0; i < 8; ++i) {
        cw += ew[i] * iw; xk[i + 1] = BOUND * (2.f * cw - 1.f);
        ch += eh[i] * ih; yk[i + 1] = BOUND * (2.f * ch - 1.f);
    }
    dk[0] = 1.f; dk[8] = 1.f;   // softplus(log(e-1)) == 1
#pragma unroll
    for (int i = 0; i < 7; ++i) {
        float v = phi[16 + i];
        dk[i + 1] = (v > 15.f) ? v : __logf(1.f + __expf(v));
    }
}

DEV void rqs_inverse(float y, const float* __restrict__ phi, float& xo, float& ldo) {
    float xk[9], yk[9], dk[9];
    rqs_knots(phi, xk, yk, dk);
    const bool inside = (y > -BOUND) && (y < BOUND);
    const float yc = fminf(fmaxf(y, -BOUND), BOUND);
    float x0 = xk[0], x1 = xk[1], y0 = yk[0], y1 = yk[1], d0 = dk[0], d1 = dk[1];
#pragma unroll
    for (int i = 1; i < 8; ++i) {
        const bool c = (yc >= yk[i]);
        x0 = c ? xk[i] : x0; x1 = c ? xk[i + 1] : x1;
        y0 = c ? yk[i] : y0; y1 = c ? yk[i + 1] : y1;
        d0 = c ? dk[i] : d0; d1 = c ? dk[i + 1] : d1;
    }
    const float w = x1 - x0, h = y1 - y0, s = h / w;
    const float tt = yc - y0, r = d0 + d1 - 2.f * s;
    const float a = h * (s - d0) + tt * r;
    const float b = h * d0 - tt * r;
    const float c = -s * tt;
    const float disc = fmaxf(b * b - 4.f * a * c, 0.f);
    float xi = 2.f * c / (-b - __builtin_sqrtf(disc));
    xi = fminf(fmaxf(xi, 0.f), 1.f);
    const float x = x0 + xi * w;
    const float xi1 = xi * (1.f - xi);
    const float den = s + r * xi1;
    const float omx = 1.f - xi;
    const float dydx = s * s * (d1 * xi * xi + 2.f * s * xi1 + d0 * omx * omx) / (den * den);
    xo  = inside ? x : y;
    ldo = inside ? -__logf(dydx) : 0.f;
}

DEV void rqs_forward(float x, const float* __restrict__ phi, float& yo, float& ldo) {
    float xk[9], yk[9], dk[9];
    rqs_knots(phi, xk, yk, dk);
    const bool inside = (x > -BOUND) && (x < BOUND);
    const float xc = fminf(fmaxf(x, -BOUND), BOUND);
    float x0 = xk[0], x1 = xk[1], y0 = yk[0], y1 = yk[1], d0 = dk[0], d1 = dk[1];
#pragma unroll
    for (int i = 1; i < 8; ++i) {
        const bool c = (xc >= xk[i]);
        x0 = c ? xk[i] : x0; x1 = c ? xk[i + 1] : x1;
        y0 = c ? yk[i] : y0; y1 = c ? yk[i + 1] : y1;
        d0 = c ? dk[i] : d0; d1 = c ? dk[i + 1] : d1;
    }
    const float w = x1 - x0, h = y1 - y0, s = h / w;
    const float xi = (xc - x0) / w;
    const float xi1 = xi * (1.f - xi);
    const float den = s + (d0 + d1 - 2.f * s) * xi1;
    const float y = y0 + h * (s * xi * xi + d0 * xi1) / den;
    const float omx = 1.f - xi;
    const float dydx = s * s * (d1 * xi * xi + 2.f * s * xi1 + d0 * omx * omx) / (den * den);
    yo  = inside ? y : x;
    ldo = inside ? __logf(dydx) : 0.f;
}

// ---------------- weight prep: masks + f16 B-fragment packing ----------------
struct PrepPtrs { const float* W[21]; };

DEV float maskv(int mt, int t, int n, int k) {
    switch (mt) {
        case 0: return 1.f;
        case 1: { int o = (t & 1) ? 31 - k : k; return (n % 31 >= o) ? 1.f : 0.f; }       // rep m1
        case 2: return (n % 31 >= k % 31) ? 1.f : 0.f;                                    // rep m2
        case 3: { int d = n / PDIM; int o = (t & 1) ? 31 - d : d;
                  return (o > k % 31) ? 1.f : 0.f; }                                      // rep m3
        case 4: { if (k >= LATENT) return 1.f;                                            // z m1 (+ctx)
                  int o = (t & 1) ? 7 - k : k; return (n % 7 >= o) ? 1.f : 0.f; }
        case 5: return (n % 7 >= k % 7) ? 1.f : 0.f;                                      // z m2
        case 6: { int d = n / PDIM; int o = (t & 1) ? 7 - d : d;
                  return (o > k % 7) ? 1.f : 0.f; }                                       // z m3
    }
    return 0.f;
}

__global__ __launch_bounds__(256) void prep_kernel(PrepPtrs P, unsigned* __restrict__ ws) {
    const int id = blockIdx.x;
    int t = 0, mt = 0, Kreal, KT, Nreal, NT, fb;
    if (id < 18) {
        t = id % 3;
        const int layer = id / 3;
        switch (layer) {
            case 0:  Kreal = 32;  KT = 1; Nreal = 128; NT = 8;  fb = t * REP_FRAGS + 0;          mt = 1; break;
            case 1:  Kreal = 128; KT = 4; Nreal = 128; NT = 8;  fb = t * REP_FRAGS + 8;          mt = 2; break;
            case 2:  Kreal = 128; KT = 4; Nreal = 736; NT = 46; fb = t * REP_FRAGS + 40;         mt = 3; break;
            case 3:  Kreal = 40;  KT = 2; Nreal = 128; NT = 8;  fb = Z_FRAG0 + t * Z_FRAGS;      mt = 4; break;
            case 4:  Kreal = 128; KT = 4; Nreal = 128; NT = 8;  fb = Z_FRAG0 + t * Z_FRAGS + 16; mt = 5; break;
            default: Kreal = 128; KT = 4; Nreal = 184; NT = 12; fb = Z_FRAG0 + t * Z_FRAGS + 48; mt = 6; break;
        }
    } else {
        switch (id) {
            case 18: Kreal = 8;   KT = 1; Nreal = 128; NT = 8; fb = ENC_W1; break;
            case 19: Kreal = 128; KT = 4; Nreal = 128; NT = 8; fb = ENC_W2; break;
            default: Kreal = 128; KT = 4; Nreal = 32;  NT = 2; fb = ENC_W3; break;
        }
        mt = 0;
    }
    const float* __restrict__ W = P.W[id];
    const int total = KT * NT * FRAG_DW;
    for (int idx = threadIdx.x; idx < total; idx += 256) {
        const int frag = idx >> 8, dw = idx & 255;
        const int lane = dw >> 3, j = dw & 7;
        const int kt = frag % KT, nt = frag / KT;
        const int n = nt * 16 + (lane & 15);
        const int k0 = kt * 32 + ((j >> 2) * 16) + ((lane < 16) ? 0 : 8) + 2 * (j & 3);
        float v0 = 0.f, v1 = 0.f;
        if (n < Nreal) {
            if (k0 < Kreal)     v0 = W[n * Kreal + k0]     * maskv(mt, t, n, k0);
            if (k0 + 1 < Kreal) v1 = W[n * Kreal + k0 + 1] * maskv(mt, t, n, k0 + 1);
        }
        union { _Float16 f[2]; unsigned u; } pk;
        pk.f[0] = (_Float16)v0;
        pk.f[1] = (_Float16)v1;
        ws[(size_t)(fb + frag) * FRAG_DW + dw] = pk.u;
    }
}

// ---------------- rep flow: 3 MAF-inverse transforms, 32-step scan ----------------
// 4 waves cooperate on one 16-row M-tile; N-tiles and RQS pairs are wave-split.
__global__ __launch_bounds__(128) void rep_flow_kernel(const float* __restrict__ xin,
                                                       unsigned* __restrict__ ws) {
    __shared__ alignas(16) float     yb[16 * 32];
    __shared__ alignas(16) float     xb[16 * 32];
    __shared__ alignas(16) _Float16  xh[16 * 32];
    __shared__ alignas(16) _Float16  h1[16 * 128];
    __shared__ alignas(16) _Float16  h2[16 * 128];
    __shared__ alignas(16) float     phib[16 * 736];
    __shared__ float ldpart[128];

    const int tid  = threadIdx.x;
    const int row0 = blockIdx.x * 16;
    float* wsf = reinterpret_cast<float*>(ws);

    for (int i = tid; i < 512; i += 128) yb[i] = xin[row0 * 32 + i];
    ldpart[tid] = 0.f;

    for (int t = 0; t < 3; ++t) {
        const unsigned* wb = ws + (size_t)t * REP_FRAGS * FRAG_DW;
        const float* b1 = wsf + REP_B0 + t * 992;
        const float* b2 = b1 + 128;
        const float* b3 = b1 + 256;
        for (int i = tid; i < 512; i += 128) xb[i] = 0.f;
        __syncthreads();
        for (int it = 0; it < 32; ++it) {
            for (int i = tid; i < 512; i += 128) xh[i] = (_Float16)xb[i];
            __syncthreads();
            mlp_layer_h<1, 8, 4, 1>(xh, 32, h1, 128, 0, wb,                 b1, 128, tid);
            __syncthreads();
            mlp_layer_h<4, 8, 4, 1>(h1, 128, h2, 128, 0, wb + 8 * FRAG_DW,  b2, 128, tid);
            __syncthreads();
            mlp_layer_f<4, 46, 4>(h2, 128, phib, 736,    wb + 40 * FRAG_DW, b3, 736, tid);
            __syncthreads();
            // 512 (row, dim) pairs over 128 threads: 4 dims each.
            const int r  = tid & 15;
            const int d0 = (tid >> 4) * 4;
            float lds = 0.f;
            for (int dd = 0; dd < 4; ++dd) {
                const int d = d0 + dd;
                float xo, ldv;
                rqs_inverse(yb[r * 32 + d], &phib[r * 736 + d * PDIM], xo, ldv);
                xb[r * 32 + d] = xo;
                lds += ldv;
            }
            if (it == 31) ldpart[tid] += lds;
            __syncthreads();
        }
        for (int i = tid; i < 512; i += 128) yb[i] = xb[i];
        __syncthreads();
    }
    for (int i = tid; i < 512; i += 128) wsf[U_OFF + row0 * 32 + i] = yb[i];
    __syncthreads();
    if (tid < 16) {
        float ld = 0.f;
#pragma unroll
        for (int j = 0; j < 8; ++j) ld += ldpart[tid + 16 * j];
        wsf[LD_OFF + row0 + tid] = ld;
    }
}

// ---------------- z flow: encoder + 3 conditional RQS transforms + logprob ----------------
__global__ __launch_bounds__(32) void z_flow_kernel(const float* __restrict__ theta,
                                                    unsigned* __restrict__ ws,
                                                    float* __restrict__ out) {
    __shared__ alignas(16) _Float16 th[16 * 32];
    __shared__ alignas(16) _Float16 inb[16 * 64];   // [z(8) | ctx(32) | zero-pad(24)]
    __shared__ alignas(16) _Float16 h1[16 * 128];
    __shared__ alignas(16) _Float16 h2[16 * 128];
    __shared__ alignas(16) float    phib[16 * 192]; // 184 padded to 192
    __shared__ float zb[16 * 8];
    __shared__ float ldz[32];

    const int lane = threadIdx.x;
    const int row0 = blockIdx.x * 16;
    float* wsf = reinterpret_cast<float*>(ws);

    for (int i = lane; i < 512; i += 32) {
        const int r = i >> 5, k = i & 31;
        th[i] = (k < 8) ? (_Float16)theta[(row0 + r) * 8 + k] : (_Float16)0.f;
    }
    for (int i = lane; i < 1024; i += 32) inb[i] = (_Float16)0.f;
    for (int i = lane; i < 128; i += 32) {
        const int r = i >> 3, d = i & 7;
        zb[i] = wsf[U_OFF + (row0 + r) * 32 + d];
    }
    ldz[lane] = 0.f;
    __syncthreads();

    // encoder (SiLU, SiLU, linear -> ctx written into inb cols 8..39 as f16)
    mlp_layer_h<1, 8, 1, 2>(th, 32, h1, 128, 0, ws + (size_t)ENC_W1 * FRAG_DW, wsf + EB0,       128, lane);
    __syncthreads();
    mlp_layer_h<4, 8, 1, 2>(h1, 128, h2, 128, 0, ws + (size_t)ENC_W2 * FRAG_DW, wsf + EB0 + 128, 128, lane);
    __syncthreads();
    mlp_layer_h<4, 2, 1, 0>(h2, 128, inb, 64, 8, ws + (size_t)ENC_W3 * FRAG_DW, wsf + EB0 + 256, 32, lane);
    __syncthreads();

    for (int t = 0; t < 3; ++t) {
        for (int i = lane; i < 128; i += 32) {
            const int r = i >> 3, d = i & 7;
            inb[r * 64 + d] = (_Float16)zb[i];
        }
        __syncthreads();
        const unsigned* wb = ws + (size_t)(Z_FRAG0 + t * Z_FRAGS) * FRAG_DW;
        const float* b1 = wsf + ZB0 + t * 440;
        const float* b2 = b1 + 128;
        const float* b3 = b1 + 256;
        mlp_layer_h<2, 8, 1, 1>(inb, 64, h1, 128, 0, wb,                 b1, 128, lane);
        __syncthreads();
        mlp_layer_h<4, 8, 1, 1>(h1, 128, h2, 128, 0, wb + 16 * FRAG_DW,  b2, 128, lane);
        __syncthreads();
        mlp_layer_f<4, 12, 1>(h2, 128, phib, 192,    wb + 48 * FRAG_DW,  b3, 184, lane);
        __syncthreads();
        const int r  = lane & 15;
        const int d0 = (lane >> 4) * 4;
        float l = 0.f;
        for (int dd = 0; dd < 4; ++dd) {
            const int d = d0 + dd;
            float yo, ldv;
            rqs_forward(zb[r * 8 + d], &phib[r * 192 + d * PDIM], yo, ldv);
            zb[r * 8 + d] = yo;
            l += ldv;
        }
        ldz[lane] += l;
        __syncthreads();
    }

    if (lane < 16) {
        const int r = lane;
        float sz = 0.f;
        for (int d = 0; d < 8; ++d) { const float v = zb[r * 8 + d]; sz += v * v; }
        float se = 0.f;
        for (int k = 8; k < 32; ++k) { const float v = wsf[U_OFF + (row0 + r) * 32 + k]; se += v * v; }
        const float lz = ldz[r] + ldz[r + 16];
        const float lp = -0.5f * sz - 0.5f * 8.f * L2PI + lz
                         - 0.5f * se - 0.5f * 24.f * L2PI
                         + wsf[LD_OFF + row0 + r];
        out[row0 + r] = lp;
    }
}

// ---------------- host launch ----------------
// d_in flat order assumption (jax pytree: dict keys sorted, so W1,W2,W3,b1,b2,b3):
//  0 x_norm, 1 theta,
//  2..7   rep[0] W1,W2,W3,b1,b2,b3     8..13 rep[1]     14..19 rep[2]
//  20..25 z[0]                         26..31 z[1]      32..37 z[2]
//  38..43 enc W1,W2,W3,b1,b2,b3
extern "C" void kernel_launch(void* const* d_in, const int* in_sizes, int n_in,
                              void* d_out, int out_size, void* d_ws, size_t ws_size,
                              hipStream_t stream) {
    (void)in_sizes; (void)n_in; (void)out_size; (void)ws_size;
    unsigned* ws = reinterpret_cast<unsigned*>(d_ws);
    float* wsf = reinterpret_cast<float*>(d_ws);

    // stage biases into workspace (graph-capture-safe D2D async copies)
    for (int t = 0; t < 3; ++t) {
        hipMemcpyAsync(wsf + REP_B0 + t * 992,       d_in[5 + 6 * t], 128 * sizeof(float), hipMemcpyDeviceToDevice, stream);
        hipMemcpyAsync(wsf + REP_B0 + t * 992 + 128, d_in[6 + 6 * t], 128 * sizeof(float), hipMemcpyDeviceToDevice, stream);
        hipMemcpyAsync(wsf + REP_B0 + t * 992 + 256, d_in[7 + 6 * t], 736 * sizeof(float), hipMemcpyDeviceToDevice, stream);
        hipMemcpyAsync(wsf + ZB0 + t * 440,          d_in[23 + 6 * t], 128 * sizeof(float), hipMemcpyDeviceToDevice, stream);
        hipMemcpyAsync(wsf + ZB0 + t * 440 + 128,    d_in[24 + 6 * t], 128 * sizeof(float), hipMemcpyDeviceToDevice, stream);
        hipMemcpyAsync(wsf + ZB0 + t * 440 + 256,    d_in[25 + 6 * t], 184 * sizeof(float), hipMemcpyDeviceToDevice, stream);
    }
    hipMemcpyAsync(wsf + EB0,       d_in[41], 128 * sizeof(float), hipMemcpyDeviceToDevice, stream);
    hipMemcpyAsync(wsf + EB0 + 128, d_in[42], 128 * sizeof(float), hipMemcpyDeviceToDevice, stream);
    hipMemcpyAsync(wsf + EB0 + 256, d_in[43], 32 * sizeof(float),  hipMemcpyDeviceToDevice, stream);

    PrepPtrs P;
    for (int t = 0; t < 3; ++t) {
        P.W[0 + t]  = (const float*)d_in[2 + 6 * t];
        P.W[3 + t]  = (const float*)d_in[3 + 6 * t];
        P.W[6 + t]  = (const float*)d_in[4 + 6 * t];
        P.W[9 + t]  = (const float*)d_in[20 + 6 * t];
        P.W[12 + t] = (const float*)d_in[21 + 6 * t];
        P.W[15 + t] = (const float*)d_in[22 + 6 * t];
    }
    P.W[18] = (const float*)d_in[38];
    P.W[19] = (const float*)d_in[39];
    P.W[20] = (const float*)d_in[40];

    prep_kernel<<<21, 256, 0, stream>>>(P, ws);
    rep_flow_kernel<<<BATCH / 16, 128, 0, stream>>>((const float*)d_in[0], ws);
    z_flow_kernel<<<BATCH / 16, 32, 0, stream>>>((const float*)d_in[1], ws, (float*)d_out);
}
